// MultimodalDecoderLayer7_73023033966621
// MI455X (gfx1250) — matrix operand order repaired and checked by reading, not compiled, not run on hardware
//
#include <hip/hip_runtime.h>
#include <hip/hip_bf16.h>
#include <math.h>

// ---------------------------------------------------------------------------
// Types for CDNA5 WMMA (wave32, V_WMMA_F32_16X16X32_BF16)
// ---------------------------------------------------------------------------
typedef __attribute__((ext_vector_type(16))) __bf16 v16bf;
typedef __attribute__((ext_vector_type(4)))  __bf16 v4bf;
typedef __attribute__((ext_vector_type(8)))  float  v8f;
typedef __attribute__((ext_vector_type(4)))  float  v4f;

__device__ __forceinline__ v4bf cvt4(v4f f) {
  v4bf h;
  h.x = (__bf16)f.x; h.y = (__bf16)f.y; h.z = (__bf16)f.z; h.w = (__bf16)f.w;
  return h;
}

// Wave-relative LDS byte offset of a generic pointer to __shared__ memory.
// Shared aperture occupies addr[63:32]; addr[31:0] is the LDS offset (ISA 10.2).
__device__ __forceinline__ unsigned lds_off(const void* p) {
  return (unsigned)(size_t)p;
}
// Async 16B copy global -> LDS (GLOBAL_LOAD_ASYNC_TO_LDS_B128, ASYNCcnt).
__device__ __forceinline__ void async_copy16(unsigned lds, const void* g) {
  asm volatile("global_load_async_to_lds_b128 %0, %1, off"
               :: "v"(lds), "v"((unsigned long long)(size_t)g) : "memory");
}
__device__ __forceinline__ void wait_async0() {
  asm volatile("s_wait_asynccnt 0" ::: "memory");
}

// ---------------------------------------------------------------------------
// GEMM: C[M,N] = epilogue(A[M,K] @ B[K,N] + bias[N]) (+ residual[M,N]) (relu)
//   A : bf16 row-major [M,K]
//   Wt: bf16 TRANSPOSED weights [N,K]  (so both LDS tiles are contiguous
//       row copies -> staged entirely with global_load_async_to_lds_b128)
//   C : fp32 (Cf) or bf16 (Ch) output
// Block: 256 threads = 8 waves. Tile: 128(M) x 128(N), K-step 64.
// Wave w owns rows [w*16, w*16+16) and all eight 16-col N tiles:
// 16 WMMAs per wave per barrier pair.
// Requires: M%128==0, N%128==0, K%64==0 (true for every GEMM in this layer).
// ---------------------------------------------------------------------------
__global__ __launch_bounds__(256)
void gemm_bf16_wmma(const __bf16* __restrict__ A, const __bf16* __restrict__ Wt,
                    const float* __restrict__ bias, const float* __restrict__ residual,
                    float* __restrict__ Cf, __bf16* __restrict__ Ch,
                    int M, int N, int K, int relu)
{
  __shared__ __bf16 As[128][64];    // [m][k] 16 KB
  __shared__ __bf16 BsT[128][64];   // [n][k] 16 KB (K-contiguous per column)

  const int tid  = threadIdx.x;
  const int bm   = blockIdx.y * 128;
  const int bn   = blockIdx.x * 128;
  const int wave = tid >> 5;
  const int lane = tid & 31;
  const int hl   = lane >> 4;    // lane-group (0: lanes 0-15, 1: lanes 16-31)
  const int lr   = lane & 15;

  v8f acc[8];
#pragma unroll
  for (int f = 0; f < 8; ++f) acc[f] = (v8f){0.f,0.f,0.f,0.f,0.f,0.f,0.f,0.f};

  for (int k0 = 0; k0 < K; k0 += 64) {
    // ---- stage both tiles (128 rows x 64 bf16 = 1024 16B chunks each) -----
    // pure async byte copies: no VGPR landing, no VALU, tracked by ASYNCcnt
#pragma unroll
    for (int i = 0; i < 4; ++i) {
      int idx = i * 256 + tid;               // 0..1023
      int r = idx >> 3;                      // row (8 chunks per 64-el row)
      int c = (idx & 7) * 8;                 // bf16 column of 16B chunk
      async_copy16(lds_off(&As[r][c]),  &A [(size_t)(bm + r) * K + k0 + c]);
      async_copy16(lds_off(&BsT[r][c]), &Wt[(size_t)(bn + r) * K + k0 + c]);
    }
    if (k0 + 64 < K) // gfx1250 global_prefetch_b8 for next A tile
      __builtin_prefetch(&A[(size_t)(bm + (tid >> 1)) * K + k0 + 64], 0, 1);
    wait_async0();
    __syncthreads();

#pragma unroll
    for (int ks = 0; ks < 2; ++ks) {
      // ---- A fragment (ISA 7.12.2, 16-bit A 16x32):
      //      lanes 0-15: VGPR0-3 K=0..7, VGPR4-7 K=16..23; lanes 16-31 +8
      const __bf16* ap = &As[wave * 16 + lr][ks * 32];
      v16bf av;
#pragma unroll
      for (int e = 0; e < 8; ++e) {
        av[e]     = ap[hl * 8 + e];          // contiguous 16B -> ds_load_b128
        av[8 + e] = ap[16 + hl * 8 + e];
      }
#pragma unroll
      for (int f = 0; f < 8; ++f) {
        // ---- B fragment: lane = column n, K same as A (e<8: hl*8+e, else 16+hl*8+e-8) ----
        const __bf16* bp = &BsT[f * 16 + lr][ks * 32];
        v16bf bv;
#pragma unroll
        for (int e = 0; e < 8; ++e) {
          bv[e]     = bp[hl * 8 + e];
          bv[8 + e] = bp[16 + hl * 8 + e];
        }
        acc[f] = __builtin_amdgcn_wmma_f32_16x16x32_bf16(
            false, av, false, bv, (short)0, acc[f], false, false);
      }
    }
    __syncthreads();
  }

  // ---- epilogue: C/D layout: VGPR j -> row (8*hl + j), col = lane%16 -------
#pragma unroll
  for (int f = 0; f < 8; ++f) {
    const int col = bn + f * 16 + lr;
    const float bs = bias ? bias[col] : 0.f;
#pragma unroll
    for (int j = 0; j < 8; ++j) {
      const int row = bm + wave * 16 + hl * 8 + j;
      float v = acc[f][j] + bs;
      if (relu) v = fmaxf(v, 0.f);
      if (residual) v += residual[(size_t)row * N + col];
      if (Ch) Ch[(size_t)row * N + col] = (__bf16)v;
      else    Cf[(size_t)row * N + col] = v;
    }
  }
}

// ---------------------------------------------------------------------------
// LayerNorm (reference: g*(x-mean)/(std+eps)+b, ddof=0, eps=1e-6). D=512.
// Output bf16 (consumed only as GEMM A-operand).
// ---------------------------------------------------------------------------
__global__ __launch_bounds__(256)
void layernorm_k(const float* __restrict__ x, const float* __restrict__ g,
                 const float* __restrict__ b, __bf16* __restrict__ y)
{
  const int row = blockIdx.x;
  const int tid = threadIdx.x;
  const float* xr = x + (size_t)row * 512;
  float v0 = xr[tid], v1 = xr[tid + 256];

  __shared__ float red[8];
  const int wave = tid >> 5, lane = tid & 31;

  float s = v0 + v1;
#pragma unroll
  for (int o = 16; o; o >>= 1) s += __shfl_xor(s, o, 32);
  if (!lane) red[wave] = s;
  __syncthreads();
  float tot = red[0]+red[1]+red[2]+red[3]+red[4]+red[5]+red[6]+red[7];
  const float mean = tot * (1.f / 512.f);
  __syncthreads();

  const float d0 = v0 - mean, d1 = v1 - mean;
  s = d0 * d0 + d1 * d1;
#pragma unroll
  for (int o = 16; o; o >>= 1) s += __shfl_xor(s, o, 32);
  if (!lane) red[wave] = s;
  __syncthreads();
  tot = red[0]+red[1]+red[2]+red[3]+red[4]+red[5]+red[6]+red[7];
  const float inv = 1.f / (sqrtf(tot * (1.f / 512.f)) + 1e-6f);

  __bf16* yr = y + (size_t)row * 512;
  yr[tid]       = (__bf16)(g[tid]       * d0 * inv + b[tid]);
  yr[tid + 256] = (__bf16)(g[tid + 256] * d1 * inv + b[tid + 256]);
}

// ---------------------------------------------------------------------------
// Attention for one (sequence, head) per block. H=8, dk=64, D=512.
//   q_row(q) = outer*qOS + inner*qIS + q*qQS
//   k_row(t) = outer*kOS + inner*kIS + t*kTS     (same map for V)
//   o_row(q) = outer*oOS + inner*oIS + q*oQS
// maskMode 1: mask[outer*Lq*Lk + q*Lk + t] == 0 -> -1e9 (trg_mask).
// Q/K/V fp32 (GEMM epilogue outputs); O bf16 (feeds o-projection GEMM).
// Lq <= 64, Lk <= 128.
// ---------------------------------------------------------------------------
__global__ __launch_bounds__(256)
void attention_k(const float* __restrict__ Qp, const float* __restrict__ Kp,
                 const float* __restrict__ Vp, __bf16* __restrict__ O,
                 const int* __restrict__ mask, int maskMode,
                 int innerCount, int Lq, int Lk,
                 int qOS, int qIS, int qQS,
                 int kOS, int kIS, int kTS,
                 int oOS, int oIS, int oQS)
{
  __shared__ __bf16 Qs[64 * 64];    //  8 KB
  __shared__ __bf16 Ks[128 * 64];   // 16 KB
  __shared__ __bf16 Vs[128 * 64];   // 16 KB
  __shared__ float probs[8 * 128];  //  4 KB (per-wave row)

  const int blk   = blockIdx.x;
  const int h     = blk & 7;
  const int seq   = blk >> 3;
  const int outer = seq / innerCount;
  const int inner = seq - outer * innerCount;
  const int hoff  = h * 64;

  for (int idx = threadIdx.x; idx < Lq * 16; idx += 256) {
    int q = idx >> 4, d = (idx & 15) * 4;
    int r = outer * qOS + inner * qIS + q * qQS;
    v4f f = *reinterpret_cast<const v4f*>(&Qp[(size_t)r * 512 + hoff + d]);
    *reinterpret_cast<v4bf*>(&Qs[q * 64 + d]) = cvt4(f);
  }
  for (int idx = threadIdx.x; idx < Lk * 16; idx += 256) {
    int t = idx >> 4, d = (idx & 15) * 4;
    int r = outer * kOS + inner * kIS + t * kTS;
    v4f fk = *reinterpret_cast<const v4f*>(&Kp[(size_t)r * 512 + hoff + d]);
    v4f fv = *reinterpret_cast<const v4f*>(&Vp[(size_t)r * 512 + hoff + d]);
    *reinterpret_cast<v4bf*>(&Ks[t * 64 + d]) = cvt4(fk);
    *reinterpret_cast<v4bf*>(&Vs[t * 64 + d]) = cvt4(fv);
  }
  __syncthreads();

  const int wave = threadIdx.x >> 5, lane = threadIdx.x & 31;

  for (int q = wave; q < Lq; q += 8) {
    float sc[4];
#pragma unroll
    for (int ti = 0; ti < 4; ++ti) {
      const int t = lane + 32 * ti;
      float a = -1e30f;
      if (t < Lk) {
        a = 0.f;
        for (int d = 0; d < 64; ++d)
          a += (float)Qs[q * 64 + d] * (float)Ks[t * 64 + d];
        a *= 0.125f;  // 1/sqrt(dk), dk=64
        if (maskMode && mask[(size_t)outer * Lq * Lk + q * Lk + t] == 0)
          a = -1e9f;
      }
      sc[ti] = a;
    }
    float mx = fmaxf(fmaxf(sc[0], sc[1]), fmaxf(sc[2], sc[3]));
#pragma unroll
    for (int o = 16; o; o >>= 1) mx = fmaxf(mx, __shfl_xor(mx, o, 32));

    float ex[4], sum = 0.f;
#pragma unroll
    for (int ti = 0; ti < 4; ++ti) {
      const int t = lane + 32 * ti;
      ex[ti] = (t < Lk) ? __expf(sc[ti] - mx) : 0.f;
      sum += ex[ti];
    }
#pragma unroll
    for (int o = 16; o; o >>= 1) sum += __shfl_xor(sum, o, 32);
    const float inv = 1.f / sum;
#pragma unroll
    for (int ti = 0; ti < 4; ++ti) {
      const int t = lane + 32 * ti;
      if (t < Lk) probs[wave * 128 + t] = ex[ti] * inv;
    }
    // cross-lane handoff through LDS within this wave (CDNA5 split counter)
    asm volatile("s_wait_dscnt 0" ::: "memory");

    const int orow = outer * oOS + inner * oIS + q * oQS;
#pragma unroll
    for (int d = lane; d < 64; d += 32) {
      float a = 0.f;
      for (int t = 0; t < Lk; ++t)
        a += probs[wave * 128 + t] * (float)Vs[t * 64 + d];
      O[(size_t)orow * 512 + hoff + d] = (__bf16)a;
    }
  }
}

// ---------------------------------------------------------------------------
// Conversion / elementwise helpers
// ---------------------------------------------------------------------------
__global__ void cvt_bf16_k(const float* __restrict__ in, __bf16* __restrict__ out,
                           size_t n4)  // n4 = elems/4
{
  size_t i = (size_t)blockIdx.x * 256 + threadIdx.x;
  if (i >= n4) return;
  reinterpret_cast<v4bf*>(out)[i] = cvt4(reinterpret_cast<const v4f*>(in)[i]);
}

// fp32 [K,N] -> bf16 [N,K] tiled transpose (K,N multiples of 32)
__global__ __launch_bounds__(256)
void cvt_tr_k(const float* __restrict__ src, __bf16* __restrict__ dst, int K, int N)
{
  __shared__ __bf16 tile[32][33];
  const int n0 = blockIdx.x * 32, k0 = blockIdx.y * 32;
  const int tx = threadIdx.x & 31, ty = threadIdx.x >> 5;   // 32x8
#pragma unroll
  for (int i = 0; i < 4; ++i) {
    int k = ty + i * 8;
    tile[k][tx] = (__bf16)src[(size_t)(k0 + k) * N + n0 + tx];
  }
  __syncthreads();
#pragma unroll
  for (int i = 0; i < 4; ++i) {
    int n = ty + i * 8;
    dst[(size_t)(n0 + n) * K + k0 + tx] = tile[tx][n];
  }
}

// out_h[((b*inner + i)*64 + q)*512 + d] = in[...] + mm[(b*64 + q)*512 + d]
__global__ void add_bcast_h_k(const float* __restrict__ in, const float* __restrict__ mm,
                              __bf16* __restrict__ out, int inner, size_t n)
{
  size_t i = (size_t)blockIdx.x * 256 + threadIdx.x;
  if (i >= n) return;
  const int d = (int)(i & 511);
  const size_t r = i >> 9;
  const int q = (int)(r & 63);
  const int b = (int)(r >> 6) / inner;
  out[i] = (__bf16)(in[i] + mm[(((size_t)b * 64 + q) << 9) + d]);
}

__global__ void add2_k(float* __restrict__ o, const float* __restrict__ a,
                       const float* __restrict__ b, size_t n)
{
  size_t i = (size_t)blockIdx.x * 256 + threadIdx.x;
  if (i < n) o[i] = a[i] + b[i];
}

// ---------------------------------------------------------------------------
// Orchestration
// ---------------------------------------------------------------------------
extern "C" void kernel_launch(void* const* d_in, const int* in_sizes, int n_in,
                              void* d_out, int out_size, void* d_ws, size_t ws_size,
                              hipStream_t stream)
{
  (void)in_sizes; (void)n_in; (void)out_size; (void)ws_size;

  const float* x    = (const float*)d_in[0];
  const float* vft  = (const float*)d_in[1];   // [16,64,49,512] flat
  const float* his  = (const float*)d_in[2];   // [16,128,512]
  const float* cap  = (const float*)d_in[3];   // [16,64,512]
  const float* qry  = (const float*)d_in[4];   // [16,32,512]
  const int*   trgm = (const int*)d_in[5];     // [16,64,64]
  const float* aw   = (const float*)d_in[10];  // [8,4,512,512]
  const float* ab   = (const float*)d_in[11];  // [8,4,512]
  const float* fw1  = (const float*)d_in[12];  // [3,512,2048]
  const float* fb1  = (const float*)d_in[13];
  const float* fw2  = (const float*)d_in[14];  // [3,2048,512]
  const float* fb2  = (const float*)d_in[15];
  const float* lg   = (const float*)d_in[16];  // [11,512]
  const float* lb   = (const float*)d_in[17];
  float* outp = (float*)d_out;

  const int Bn = 16, Ttrg = 64, T = 64, S = 49, D = 512, DFF = 2048, Hh = 8;
  const int N0 = Bn * Ttrg;                    // 1024
  const int MV = Bn * T * S;                   // 50176 (vft rows)
  const int MS = Bn * T * Ttrg;                // 65536 (s_out rows)

  char* ws = (char*)d_ws;
  size_t off = 0;
  auto allocF = [&](size_t elems) -> float* {
    float* p = (float*)(ws + off);
    off += ((elems * 4) + 255) & ~(size_t)255;
    return p;
  };
  auto allocH = [&](size_t elems) -> __bf16* {
    __bf16* p = (__bf16*)(ws + off);
    off += ((elems * 2) + 255) & ~(size_t)255;
    return p;
  };
  // fp32
  float*  qbuf  = allocF((size_t)N0 * D);
  float*  kbuf  = allocF((size_t)2048 * D);
  float*  vbuf  = allocF((size_t)2048 * D);
  float*  bufA  = allocF((size_t)N0 * D);
  float*  bufB  = allocF((size_t)N0 * D);
  float*  mm    = allocF((size_t)N0 * D);
  float*  tsb   = allocF((size_t)N0 * D);
  float*  stb   = allocF((size_t)N0 * D);
  float*  preb  = allocF((size_t)N0 * D);
  float*  S1    = allocF((size_t)MS * D);      // 128 MiB fp32 slab
  float*  S2    = allocF((size_t)MS * D);      // 128 MiB fp32 slab
  // bf16
  __bf16* tlnH  = allocH((size_t)N0 * D);
  __bf16* attOH = allocH((size_t)N0 * D);
  __bf16* ffbH  = allocH((size_t)N0 * DFF);
  __bf16* hisH  = allocH((size_t)2048 * D);
  __bf16* capH  = allocH((size_t)1024 * D);
  __bf16* qryH  = allocH((size_t)512 * D);
  __bf16* vftH  = allocH((size_t)MV * D);
  __bf16* Hbig  = allocH((size_t)MS * D);      // 64 MiB bf16 slab
  __bf16* wAt   = allocH((size_t)32 * D * D);  // attn weights, [N,K] bf16
  __bf16* wF1t  = allocH((size_t)3 * DFF * D);
  __bf16* wF2t  = allocH((size_t)3 * D * DFF);

  auto gemm = [&](const __bf16* A, const __bf16* Wt, const float* bias,
                  const float* res, float* Cf, __bf16* Ch,
                  int M, int N, int K, int relu) {
    dim3 g(N / 128, M / 128);
    gemm_bf16_wmma<<<g, 256, 0, stream>>>(A, Wt, bias, res, Cf, Ch, M, N, K, relu);
  };
  auto lnorm = [&](const float* in, int li, __bf16* out) {
    layernorm_k<<<N0, 256, 0, stream>>>(in, lg + (size_t)li * D, lb + (size_t)li * D, out);
  };
  auto attn = [&](const float* Qp, const float* Kp, const float* Vp, __bf16* O,
                  const int* mask, int mmode, int numSeq, int inner, int Lq, int Lk,
                  int qOS, int qIS, int qQS, int kOS, int kIS, int kTS,
                  int oOS, int oIS, int oQS) {
    attention_k<<<numSeq * Hh, 256, 0, stream>>>(Qp, Kp, Vp, O, mask, mmode,
        inner, Lq, Lk, qOS, qIS, qQS, kOS, kIS, kTS, oOS, oIS, oQS);
  };
  auto AWT = [&](int i, int j) { return wAt + ((size_t)i * 4 + j) * D * D; };
  auto AB  = [&](int i, int j) { return ab + ((size_t)i * 4 + j) * D; };

  // ===== one-time: weights -> bf16 [N,K]; big activations -> bf16 ==========
  for (int m = 0; m < 32; ++m)
    cvt_tr_k<<<dim3(D / 32, D / 32), 256, 0, stream>>>(
        aw + (size_t)m * D * D, wAt + (size_t)m * D * D, D, D);
  for (int m = 0; m < 3; ++m) {
    cvt_tr_k<<<dim3(DFF / 32, D / 32), 256, 0, stream>>>(
        fw1 + (size_t)m * D * DFF, wF1t + (size_t)m * DFF * D, D, DFF);
    cvt_tr_k<<<dim3(D / 32, DFF / 32), 256, 0, stream>>>(
        fw2 + (size_t)m * DFF * D, wF2t + (size_t)m * D * DFF, DFF, D);
  }
  cvt_bf16_k<<<(2048 * D / 4 + 255) / 256, 256, 0, stream>>>(his, hisH, 2048 * D / 4);
  cvt_bf16_k<<<(1024 * D / 4 + 255) / 256, 256, 0, stream>>>(cap, capH, 1024 * D / 4);
  cvt_bf16_k<<<(512  * D / 4 + 255) / 256, 256, 0, stream>>>(qry, qryH, 512 * D / 4);
  cvt_bf16_k<<<(unsigned)(((size_t)MV * D / 4 + 255) / 256), 256, 0, stream>>>(
      vft, vftH, (size_t)MV * D / 4);

  // ===== Stage A: 4 sequential MHAs on [B,Ttrg,D] ==========================
  const float* cur = x;
  float* nxt[4] = { bufA, bufB, bufA, mm };
  for (int i = 0; i < 4; ++i) {
    lnorm(cur, i, tlnH);
    gemm(tlnH, AWT(i, 0), AB(i, 0), nullptr, qbuf, nullptr, N0, D, D, 0);
    const __bf16* srcH; int Mk;
    if (i == 0)      { srcH = tlnH; Mk = N0; }       // self-attn
    else if (i == 1) { srcH = hisH; Mk = Bn * 128; }
    else if (i == 2) { srcH = capH; Mk = Bn * 64; }
    else             { srcH = qryH; Mk = Bn * 32; }
    gemm(srcH, AWT(i, 1), AB(i, 1), nullptr, kbuf, nullptr, Mk, D, D, 0);
    gemm(srcH, AWT(i, 2), AB(i, 2), nullptr, vbuf, nullptr, Mk, D, D, 0);
    const int Lk = Mk / Bn;
    attn(qbuf, kbuf, vbuf, attOH, (i == 0) ? trgm : nullptr, (i == 0) ? 1 : 0,
         Bn, 1, Ttrg, Lk,
         Ttrg, 0, 1,   Lk, 0, 1,   Ttrg, 0, 1);
    gemm(attOH, AWT(i, 3), AB(i, 3), cur, nxt[i], nullptr, N0, D, D, 0);
    cur = nxt[i];
  }
  // cur == mm

  // ===== Stage B: temporal2spatial =========================================
  lnorm(mm, 4, tlnH);
  gemm(tlnH, AWT(4, 0), AB(4, 0), nullptr, qbuf, nullptr, N0, D, D, 0);   // q4
  gemm(vftH, AWT(4, 1), AB(4, 1), nullptr, S1, nullptr, MV, D, D, 0);     // k4
  gemm(vftH, AWT(4, 2), AB(4, 2), nullptr, S2, nullptr, MV, D, D, 0);     // v4
  // mha4: seq=(b,s); keys t -> vft row b*T*S + s + t*S; out row b*S*64 + s*64 + q
  attn(qbuf, S1, S2, Hbig, nullptr, 0, Bn * S, S, Ttrg, T,
       Ttrg, 0, 1,   T * S, 1, S,   S * Ttrg, Ttrg, 1);
  gemm(Hbig, AWT(4, 3), AB(4, 3), nullptr, S1, nullptr, MV, D, D, 0);     // t_out
  {
    size_t n = (size_t)MV * D;
    add_bcast_h_k<<<(unsigned)((n + 255) / 256), 256, 0, stream>>>(S1, mm, Hbig, S, n);
  }
  lnorm(mm, 5, tlnH);
  gemm(tlnH, AWT(5, 0), AB(5, 0), nullptr, qbuf, nullptr, N0, D, D, 0);   // qp5
  gemm(Hbig, AWT(5, 1), AB(5, 1), nullptr, S2, nullptr, MV, D, D, 0);     // kp5
  gemm(Hbig, AWT(5, 2), AB(5, 2), nullptr, S1, nullptr, MV, D, D, 0);     // vp5
  // mha5: seq=(b,qhat), Lq=1; keys s -> row b*S*64 + qhat + s*64
  attn(qbuf, S2, S1, attOH, nullptr, 0, Bn * Ttrg, Ttrg, 1, S,
       Ttrg, 1, 0,   S * Ttrg, 1, Ttrg,   Ttrg, 1, 0);
  gemm(attOH, AWT(5, 3), AB(5, 3), mm, tsb, nullptr, N0, D, D, 0);        // ts
  // ffn0 (ln 6)
  lnorm(tsb, 6, tlnH);
  gemm(tlnH, wF1t + 0 * (size_t)DFF * D, fb1 + 0 * DFF, nullptr, nullptr, ffbH,
       N0, DFF, D, 1);
  gemm(ffbH, wF2t + 0 * (size_t)D * DFF, fb2 + 0 * D, tsb, tsb, nullptr,
       N0, D, DFF, 0);

  // ===== Stage C: spatial2temporal =========================================
  gemm(vftH, AWT(6, 1), AB(6, 1), nullptr, S1, nullptr, MV, D, D, 0);     // kp6
  gemm(vftH, AWT(6, 2), AB(6, 2), nullptr, S2, nullptr, MV, D, D, 0);     // vp6
  lnorm(mm, 7, tlnH);
  gemm(tlnH, AWT(6, 0), AB(6, 0), nullptr, qbuf, nullptr, N0, D, D, 0);   // q6
  // mha6: seq=(b,t); keys s -> vft row b*T*S + t*S + s; out row b*T*64 + t*64 + q
  attn(qbuf, S1, S2, Hbig, nullptr, 0, Bn * T, T, Ttrg, S,
       Ttrg, 0, 1,   T * S, S, 1,   T * Ttrg, Ttrg, 1);
  gemm(Hbig, AWT(6, 3), AB(6, 3), nullptr, S1, nullptr, MS, D, D, 0);     // s_out
  {
    size_t n = (size_t)MS * D;
    add_bcast_h_k<<<(unsigned)((n + 255) / 256), 256, 0, stream>>>(S1, mm, Hbig, T, n);
  }
  lnorm(mm, 8, tlnH);
  gemm(tlnH, AWT(7, 0), AB(7, 0), nullptr, qbuf, nullptr, N0, D, D, 0);   // qp7
  gemm(Hbig, AWT(7, 1), AB(7, 1), nullptr, S2, nullptr, MS, D, D, 0);     // kp7
  gemm(Hbig, AWT(7, 2), AB(7, 2), nullptr, S1, nullptr, MS, D, D, 0);     // vp7
  // mha7: seq=(b,qhat), Lq=1; keys t -> row b*T*64 + qhat + t*64
  attn(qbuf, S2, S1, attOH, nullptr, 0, Bn * Ttrg, Ttrg, 1, T,
       Ttrg, 1, 0,   T * Ttrg, 1, Ttrg,   Ttrg, 1, 0);
  gemm(attOH, AWT(7, 3), AB(7, 3), mm, stb, nullptr, N0, D, D, 0);        // st
  // ffn1 (ln 9)
  lnorm(stb, 9, tlnH);
  gemm(tlnH, wF1t + 1 * (size_t)DFF * D, fb1 + 1 * DFF, nullptr, nullptr, ffbH,
       N0, DFF, D, 1);
  gemm(ffbH, wF2t + 1 * (size_t)D * DFF, fb2 + 1 * D, stb, stb, nullptr,
       N0, D, DFF, 0);

  // ===== out = (ts + st) + ffn2(LN10(ts + st)) =============================
  {
    size_t n = (size_t)N0 * D;
    add2_k<<<(unsigned)((n + 255) / 256), 256, 0, stream>>>(preb, tsb, stb, n);
  }
  lnorm(preb, 10, tlnH);
  gemm(tlnH, wF1t + 2 * (size_t)DFF * D, fb1 + 2 * DFF, nullptr, nullptr, ffbH,
       N0, DFF, D, 1);
  gemm(ffbH, wF2t + 2 * (size_t)D * DFF, fb2 + 2 * D, preb, outp, nullptr,
       N0, D, DFF, 0);
}
